// LanguageConditionedSegmentation_49469433315719
// MI455X (gfx1250) — matrix operand order repaired and checked by reading, not compiled, not run on hardware
//
#include <hip/hip_runtime.h>

typedef _Float16 h2   __attribute__((ext_vector_type(2)));
typedef _Float16 v16h __attribute__((ext_vector_type(16)));
typedef float    v8f  __attribute__((ext_vector_type(8)));
typedef float    v4f  __attribute__((ext_vector_type(4)));

#define N_BOX 256     // T*B = 8*32
#define BPF   32
#define HIMG  512
#define WIMG  512
#define SCR   32
#define CE    3072    // 3*32*32
#define DD    128
#define MQ    64
#define EPSF  1e-10f

// ---------------- 1) w_embed f32 -> f16, TRANSPOSED -------------------------
// wht[col][k] = w[k][col]  so B-fragment K-pairs are contiguous h2 loads
__global__ void k_wconv(const float* __restrict__ w, _Float16* __restrict__ wht, int n) {
  int i = blockIdx.x * blockDim.x + threadIdx.x;   // i = k*128 + col (coalesced read)
  if (i < n) {
    int k = i >> 7, col = i & 127;
    wht[(size_t)col * CE + k] = (_Float16)w[i];
  }
}

// ---------------- 2) crop + standardize -------------------------------------
// one block (256 thr) per box; 12 elements per thread kept in registers
__global__ void k_crop(const float* __restrict__ images, const int* __restrict__ boxes,
                       float* __restrict__ cropf, _Float16* __restrict__ croph) {
  __shared__ float ssum[256], ssq[256];
  const int n = blockIdx.x, tid = threadIdx.x;
  const int t = n / BPF;
  const int x0 = boxes[n*4+0], y0 = boxes[n*4+1];
  const int w  = boxes[n*4+2] - x0, h = boxes[n*4+3] - y0;
  float vals[12], s = 0.f, s2 = 0.f;
#pragma unroll
  for (int i = 0; i < 12; ++i) {
    int e  = tid + 256*i;
    int c  = e >> 10, rem = e & 1023;
    int j  = rem >> 5, ii = rem & 31;
    int iy = y0 + ((j * h) >> 5);
    int ix = x0 + ((ii * w) >> 5);
    float v = images[(((size_t)t*3 + c)*HIMG + iy)*WIMG + ix];
    vals[i] = v; s += v; s2 += v*v;
  }
  ssum[tid] = s; ssq[tid] = s2; __syncthreads();
  for (int off = 128; off > 0; off >>= 1) {
    if (tid < off) { ssum[tid] += ssum[tid+off]; ssq[tid] += ssq[tid+off]; }
    __syncthreads();
  }
  const float mean = ssum[0] * (1.f/CE);
  const float var  = fmaxf(ssq[0] * (1.f/CE) - mean*mean, 0.f);
  const float inv  = 1.f / (sqrtf(var) + 1e-5f);
#pragma unroll
  for (int i = 0; i < 12; ++i) {
    int e = tid + 256*i;
    float z = (vals[i] - mean) * inv;
    cropf[(size_t)n*CE + e] = z;
    croph[(size_t)n*CE + e] = (_Float16)z;
  }
}

// ---------------- 3) embed GEMM via v_wmma_f32_16x16x32_f16 -----------------
// C[256,128] = A[256,3072] * W[3072,128]; 1 wave per 16x16 tile, 128 waves.
// Wt is W transposed ([128][3072]) so both fragments load as contiguous h2.
__global__ void k_gemm(const _Float16* __restrict__ A, const _Float16* __restrict__ Wt,
                       float* __restrict__ C) {
  const int gid   = blockIdx.x * blockDim.x + threadIdx.x;
  const int wave  = gid >> 5;
  const int lane  = gid & 31;
  const int tileM = wave >> 3;        // 0..15
  const int tileN = wave & 7;         // 0..7
  const int half  = lane >> 4;        // 0 or 1
  const int mr    = lane & 15;
  const _Float16* __restrict__ arow = A  + (size_t)(tileM*16 + mr) * CE; // A: lane=row M
  const _Float16* __restrict__ bcol = Wt + (size_t)(tileN*16 + mr) * CE; // B: lane=col N
  v8f c = {};
  for (int kk = 0; kk < CE; kk += 32) {
    v16h a, b;
#pragma unroll
    for (int v8 = 0; v8 < 8; ++v8) {
      // A 16-bit 16x32 layout: VGPR v -> K pair, halves interleave 0-7 / 8-15
      int kb = 2*v8 + (v8 >= 4 ? 8 : 0) + 8*half;
      h2 pa = *(const h2*)(arow + kk + kb);
      a[2*v8]   = pa[0];
      a[2*v8+1] = pa[1];
      // B 16-bit 32x16 layout: same K order as A (halves interleave 0-7 / 8-15)
      int k0 = 2*v8 + (v8 >= 4 ? 8 : 0) + 8*half;
      h2 pb = *(const h2*)(bcol + kk + k0);
      b[2*v8]   = pb[0];
      b[2*v8+1] = pb[1];
    }
    c = __builtin_amdgcn_wmma_f32_16x16x32_f16(false, a, false, b,
                                               (short)0, c, false, false);
  }
#pragma unroll
  for (int r = 0; r < 8; ++r) {
    int m = tileM*16 + r + 8*half;    // C/D layout: lanes 16-31 hold M=r+8
    C[(size_t)m*DD + tileN*16 + mr] = c[r];
  }
}

// ---------------- 4) KDE score per crop -------------------------------------
// one block (64 thr) per row of v; thread m handles query m
__global__ void k_score(const float* __restrict__ v, const float* __restrict__ q,
                        float* __restrict__ score) {
  __shared__ float vrow[DD];
  __shared__ float red[MQ], redm[MQ];
  const int row = blockIdx.x, m = threadIdx.x;
  float a0 = v[(size_t)row*DD + m];
  float a1 = v[(size_t)row*DD + 64 + m];
  vrow[m] = a0; vrow[64 + m] = a1;
  red[m] = a0*a0 + a1*a1;
  __syncthreads();
  for (int off = 32; off > 0; off >>= 1) {
    if (m < off) red[m] += red[m + off];
    __syncthreads();
  }
  const float sv   = red[0];
  const float invV = 1.f / (sqrtf(sv) + EPSF);
  __syncthreads();
  float sq = 0.f, dot = 0.f;
  for (int d = 0; d < DD; ++d) {
    float qd = q[(size_t)m*DD + d];
    sq  += qd*qd;
    dot += vrow[d]*qd;
  }
  const float invQ = 1.f / (sqrtf(sq) + EPSF);
  const float d2   = sv*invV*invV + sq*invQ*invQ - 2.f*dot*invV*invQ;
  const float dens = expf(-d2 * 0.25f);          // / (2*KDE_VAR) = /4
  red[m] = dens; redm[m] = dens;
  __syncthreads();
  for (int off = 32; off > 0; off >>= 1) {
    if (m < off) { red[m] += red[m+off]; redm[m] = fmaxf(redm[m], redm[m+off]); }
    __syncthreads();
  }
  if (m == 0) score[row] = redm[0] / (red[0] + EPSF);
}

// ---------------- 5) conv3x3 + sigmoid, fold score --------------------------
__global__ void k_conv(const float* __restrict__ cropf, const float* __restrict__ cw,
                       const float* __restrict__ cb, const float* __restrict__ score,
                       float* __restrict__ mask) {
  __shared__ float tile[3][SCR][SCR];
  const int n = blockIdx.x, tid = threadIdx.x;
#pragma unroll
  for (int i = 0; i < 12; ++i) {
    int e = tid + 256*i;
    tile[e >> 10][(e & 1023) >> 5][e & 31] = cropf[(size_t)n*CE + e];
  }
  float wreg[27];
#pragma unroll
  for (int i = 0; i < 27; ++i) wreg[i] = cw[i];
  const float bias = cb[0];
  const float sc   = score[n];
  __syncthreads();
#pragma unroll
  for (int i = 0; i < 4; ++i) {
    int p = tid + 256*i;
    int y = p >> 5, x = p & 31;
    float acc = bias;
#pragma unroll
    for (int c = 0; c < 3; ++c)
#pragma unroll
      for (int ky = 0; ky < 3; ++ky) {
        int yy = y + ky - 1;
        if (yy < 0 || yy >= SCR) continue;
#pragma unroll
        for (int kx = 0; kx < 3; ++kx) {
          int xx = x + kx - 1;
          if (xx < 0 || xx >= SCR) continue;
          acc += wreg[(c*3 + ky)*3 + kx] * tile[c][yy][xx];
        }
      }
    mask[(size_t)n*1024 + p] = sc / (1.f + expf(-acc));
  }
}

// ---------------- 6) nearest-index lookup tables ----------------------------
__global__ void k_tables(const int* __restrict__ boxes,
                         int* __restrict__ iyt, int* __restrict__ ixt) {
  const int n = blockIdx.x, c = threadIdx.x;   // c in [0,512)
  const int x0 = boxes[n*4+0], y0 = boxes[n*4+1];
  const int w  = boxes[n*4+2] - x0, h = boxes[n*4+3] - y0;
  int ry = c - y0;
  iyt[(n << 9) + c] = (ry >= 0 && ry < h) ? min((ry * SCR) / h, SCR-1) : -1;
  int rx = c - x0;
  ixt[(n << 9) + c] = (rx >= 0 && rx < w) ? min((rx * SCR) / w, SCR-1) : -1;
}

// ---------------- 7) paste-back (bandwidth-critical) ------------------------
// 1 thread = 4 consecutive x pixels; NT 16-byte stores (268 MB > L2)
__global__ void k_paste(const float* __restrict__ mask, const int* __restrict__ iyt,
                        const int* __restrict__ ixt, float* __restrict__ out) {
  const int n  = blockIdx.x >> 8;
  const int y  = ((blockIdx.x & 255) << 1) + (threadIdx.x >> 7);
  const int xb = (threadIdx.x & 127) << 2;
  const int iy = iyt[(n << 9) + y];
  const float* mrow = mask + (size_t)n*1024 + (iy > 0 ? iy : 0)*SCR;
  v4f o;
  { int ix = ixt[(n << 9) + xb + 0]; o.x = (iy >= 0 && ix >= 0) ? mrow[ix] : 0.f; }
  { int ix = ixt[(n << 9) + xb + 1]; o.y = (iy >= 0 && ix >= 0) ? mrow[ix] : 0.f; }
  { int ix = ixt[(n << 9) + xb + 2]; o.z = (iy >= 0 && ix >= 0) ? mrow[ix] : 0.f; }
  { int ix = ixt[(n << 9) + xb + 3]; o.w = (iy >= 0 && ix >= 0) ? mrow[ix] : 0.f; }
  v4f* dst = (v4f*)(out + (((size_t)n << 18) + ((size_t)y << 9) + xb));
  __builtin_nontemporal_store(o, dst);
}

extern "C" void kernel_launch(void* const* d_in, const int* in_sizes, int n_in,
                              void* d_out, int out_size, void* d_ws, size_t ws_size,
                              hipStream_t stream) {
  const float* images = (const float*)d_in[0];
  const int*   boxes  = (const int*)  d_in[1];
  const float* wemb   = (const float*)d_in[2];
  const float* qvec   = (const float*)d_in[3];
  const float* convw  = (const float*)d_in[4];
  const float* convb  = (const float*)d_in[5];
  float* out = (float*)d_out;

  // workspace carve-up (all offsets multiples of 1024 bytes)
  char* ws = (char*)d_ws;
  float*    cropf = (float*)   (ws);                         // 256*3072 f32  = 3 MB
  _Float16* croph = (_Float16*)(ws + 3145728);               // 256*3072 f16  = 1.5 MB
  _Float16* wht   = (_Float16*)(ws + 4718592);               // 128*3072 f16  = 0.75 MB (transposed)
  float*    v     = (float*)   (ws + 5505024);               // 256*128 f32
  float*    score = (float*)   (ws + 5636096);               // 256 f32
  float*    mask  = (float*)   (ws + 5637120);               // 256*1024 f32  = 1 MB
  int*      iyt   = (int*)     (ws + 6685696);               // 256*512 i32   = 0.5 MB
  int*      ixt   = (int*)     (ws + 7209984);               // 256*512 i32   = 0.5 MB
  (void)in_sizes; (void)n_in; (void)out_size; (void)ws_size;

  k_wconv <<<(CE*DD + 255)/256, 256, 0, stream>>>(wemb, wht, CE*DD);
  k_crop  <<<N_BOX, 256, 0, stream>>>(images, boxes, cropf, croph);
  k_gemm  <<<16, 256, 0, stream>>>(croph, wht, v);           // 128 waves = 128 tiles
  k_score <<<N_BOX, MQ, 0, stream>>>(v, qvec, score);
  k_conv  <<<N_BOX, 256, 0, stream>>>(cropf, convw, convb, score, mask);
  k_tables<<<N_BOX, 512, 0, stream>>>(boxes, iyt, ixt);
  k_paste <<<N_BOX*(HIMG/2), 256, 0, stream>>>(mask, iyt, ixt, out);
}
